// Model_55009941127354
// MI455X (gfx1250) — compile-verified, matrix operand order checked by reading, not run on hardware
//
#include <hip/hip_runtime.h>
#include <hip/hip_bf16.h>
#include <math.h>

// ---------------------------------------------------------------------------
// Shapes (fixed by the reference): B=16, T=768, N=768, horizon=1, F=6
// Pipeline after algebraic simplification:
//   k,q   : GEMV
//   A     : mean_b softmax_j(leaky(k_i + q_j, .2))            (T x T)
//   M     : I - normalized Laplacian of A_sym
//   Mbar  : 2 M^3 + M^2 - 0.5 M          (collapses the poly-k mean)
//   G     : Re(DFT6 * w_freq * IDFT6)    (collapses fft->wfreq->ifft)
//   GF    : G @ fc1_w
//   Z[b]  : Mbar @ x[b]^T                (WMMA, bf16 stored directly)
//   out   : fc2( leaky(Z[b] @ GF + fc1_b) )   (WMMA, fc2 fused in epilogue)
// ---------------------------------------------------------------------------

#define ALPHA_LEAKY 0.2f
#define FC_LEAKY    0.01f
#define PI_F        3.14159265358979323846f

typedef __attribute__((ext_vector_type(16))) __bf16         v16bf;
typedef __attribute__((ext_vector_type(8)))  float          v8f;
typedef __attribute__((ext_vector_type(8)))  unsigned short ushort8;

union Frag16 { ushort8 u[2]; v16bf v; };

__device__ __forceinline__ unsigned short f32_to_bf16_rne(float f) {
    unsigned int u = __float_as_uint(f);
    u += 0x7fffu + ((u >> 16) & 1u);
    return (unsigned short)(u >> 16);
}

// Fragment load per CDNA5 ISA 7.12.2 (16-bit A 16x32): lane L<16 -> row r=L,
// K in {k0..k0+7} U {k0+16..k0+23}; lane L>=16 -> same row set, K offset +8.
// Bt is stored "column-major as rows" (Bt[n*K+k] = B[k][n]) so B fragments use
// the identical addressing.
__device__ __forceinline__ v16bf load_frag(const unsigned short* __restrict__ base,
                                           int ld, int rc, int k0, int hi) {
    Frag16 f;
    const unsigned short* p = base + (size_t)rc * ld + k0 + hi * 8;
    f.u[0] = *(const ushort8*)(p);
    f.u[1] = *(const ushort8*)(p + 16);
    return f.v;
}

// ---------------------------------------------------------------------------
// Generic bf16 WMMA GEMM: acc[M,N](f32) = A[M,K](bf16) * B[K,N], B given
// pre-transposed as Bt[N,K](bf16). Requires M%128==0, N%64==0, K%32==0.
// Epilogue modes:
//   0: store f32 to C
//   2: store bf16 to C16 (row-major [M,N] -> ready as next GEMM's A operand)
//   3: fused v = leaky(acc + bias[n], .01) * w2[n]; per-row partial sums over
//      this block-column written to part[row*(gridDim.x*2) + bx*2 + wc]
// ---------------------------------------------------------------------------
__global__ __launch_bounds__(256) void gemm_bf16_wmma(
    const unsigned short* __restrict__ A, const unsigned short* __restrict__ Bt,
    float* __restrict__ C, unsigned short* __restrict__ C16,
    float* __restrict__ part, int M, int N, int K,
    long long sA, long long sB, long long sC,
    const float* __restrict__ bias, const float* __restrict__ w2, int mode)
{
    const int bz = blockIdx.z;
    A  += (size_t)bz * sA;
    Bt += (size_t)bz * sB;

    const int wave = threadIdx.x >> 5;
    const int lane = threadIdx.x & 31;
    const int r    = lane & 15;
    const int hi   = lane >> 4;
    const int wr   = wave >> 1;            // 0..3
    const int wc   = wave & 1;             // 0..1
    const int m0   = blockIdx.y * 128 + wr * 32;
    const int n0   = blockIdx.x * 64  + wc * 32;

    v8f acc[2][2] = {};

    for (int k0 = 0; k0 < K; k0 += 32) {
        v16bf a0 = load_frag(A,  K, m0 + r,      k0, hi);
        v16bf a1 = load_frag(A,  K, m0 + 16 + r, k0, hi);
        v16bf b0 = load_frag(Bt, K, n0 + r,      k0, hi);
        v16bf b1 = load_frag(Bt, K, n0 + 16 + r, k0, hi);

        if (k0 + 32 < K) {  // global_prefetch_b8 of the next k-tile
            __builtin_prefetch(A  + (size_t)(m0 + r) * K + k0 + 32, 0, 1);
            __builtin_prefetch(Bt + (size_t)(n0 + r) * K + k0 + 32, 0, 1);
        }

        acc[0][0] = __builtin_amdgcn_wmma_f32_16x16x32_bf16(false, a0, false, b0,
                        (short)0, acc[0][0], false, false);
        acc[0][1] = __builtin_amdgcn_wmma_f32_16x16x32_bf16(false, a0, false, b1,
                        (short)0, acc[0][1], false, false);
        acc[1][0] = __builtin_amdgcn_wmma_f32_16x16x32_bf16(false, a1, false, b0,
                        (short)0, acc[1][0], false, false);
        acc[1][1] = __builtin_amdgcn_wmma_f32_16x16x32_bf16(false, a1, false, b1,
                        (short)0, acc[1][1], false, false);
    }

    // C/D layout: VGPR v -> row m0+i*16+hi*8+v, col n0+j*16+r
    if (mode == 3) {
        // fused fc1-bias + leaky + fc2 dot: partial over this wave's 32 cols
        float rowsum[2][8];
        #pragma unroll
        for (int i = 0; i < 2; ++i)
            #pragma unroll
            for (int v = 0; v < 8; ++v) rowsum[i][v] = 0.f;

        #pragma unroll
        for (int i = 0; i < 2; ++i)
          #pragma unroll
          for (int j = 0; j < 2; ++j) {
            const int nb = n0 + j * 16 + r;
            const float bn = bias[nb], wn = w2[nb];
            #pragma unroll
            for (int v = 0; v < 8; ++v) {
                float val = acc[i][j][v] + bn;
                val = val >= 0.f ? val : FC_LEAKY * val;
                rowsum[i][v] += val * wn;
            }
        }
        // reduce across the 16 lanes of each half (xor < 16 stays in-half)
        #pragma unroll
        for (int i = 0; i < 2; ++i)
            #pragma unroll
            for (int v = 0; v < 8; ++v)
                #pragma unroll
                for (int off = 1; off < 16; off <<= 1)
                    rowsum[i][v] += __shfl_xor(rowsum[i][v], off, 32);
        if (r == 0) {
            const int P = gridDim.x * 2;
            const int p = blockIdx.x * 2 + wc;
            #pragma unroll
            for (int i = 0; i < 2; ++i) {
                const int mb = m0 + i * 16 + hi * 8;
                #pragma unroll
                for (int v = 0; v < 8; ++v) {
                    const long long rowg = (long long)bz * M + mb + v;
                    part[rowg * P + p] = rowsum[i][v];
                }
            }
        }
        return;
    }

    #pragma unroll
    for (int i = 0; i < 2; ++i)
      #pragma unroll
      for (int j = 0; j < 2; ++j) {
        const int mb = m0 + i * 16 + hi * 8;
        const int nb = n0 + j * 16 + r;
        #pragma unroll
        for (int v = 0; v < 8; ++v) {
            const float val = acc[i][j][v];
            const size_t idx = (size_t)bz * sC + (size_t)(mb + v) * N + nb;
            if (mode == 2) C16[idx] = f32_to_bf16_rne(val);
            else           C[idx]   = val;
        }
    }
}

// --------------------------- support kernels --------------------------------

// k = x@wk, q = x@wq; one wave per (b,t) row.
__global__ __launch_bounds__(256) void kq_kernel(
    const float* __restrict__ x, const float* __restrict__ wk,
    const float* __restrict__ wq, float* __restrict__ kbuf,
    float* __restrict__ qbuf, int rows, int N)
{
    const int row  = blockIdx.x * 8 + (threadIdx.x >> 5);
    const int lane = threadIdx.x & 31;
    if (row >= rows) return;
    const float* xr = x + (size_t)row * N;
    float sk = 0.f, sq = 0.f;
    for (int j = lane; j < N; j += 32) {
        float v = xr[j];
        sk += v * wk[j];
        sq += v * wq[j];
    }
    for (int off = 16; off > 0; off >>= 1) {
        sk += __shfl_xor(sk, off, 32);
        sq += __shfl_xor(sq, off, 32);
    }
    if (lane == 0) { kbuf[row] = sk; qbuf[row] = sq; }
}

// A[i,:] = mean_b softmax_j(leaky(k[b,i] + q[b,j], 0.2)). Block per row i.
__global__ __launch_bounds__(256) void attn_meanA_kernel(
    const float* __restrict__ kbuf, const float* __restrict__ qbuf,
    float* __restrict__ A, int B, int T)
{
    const int i   = blockIdx.x;
    const int tid = threadIdx.x;
    __shared__ float red[256];
    float accj[3] = {0.f, 0.f, 0.f};

    for (int b = 0; b < B; ++b) {
        const float ki = kbuf[b * T + i];
        float s[3];
        float lmax = -1e30f;
        #pragma unroll
        for (int u = 0; u < 3; ++u) {
            const int j = tid + u * 256;
            float v = ki + qbuf[b * T + j];
            v = v >= 0.f ? v : ALPHA_LEAKY * v;
            s[u] = v;
            lmax = fmaxf(lmax, v);
        }
        red[tid] = lmax; __syncthreads();
        for (int off = 128; off > 0; off >>= 1) {
            if (tid < off) red[tid] = fmaxf(red[tid], red[tid + off]);
            __syncthreads();
        }
        const float m = red[0];
        __syncthreads();
        float lsum = 0.f;
        #pragma unroll
        for (int u = 0; u < 3; ++u) { s[u] = __expf(s[u] - m); lsum += s[u]; }
        red[tid] = lsum; __syncthreads();
        for (int off = 128; off > 0; off >>= 1) {
            if (tid < off) red[tid] += red[tid + off];
            __syncthreads();
        }
        const float inv = 1.f / red[0];
        __syncthreads();
        #pragma unroll
        for (int u = 0; u < 3; ++u) accj[u] += s[u] * inv;
    }
    const float invB = 1.f / (float)B;
    #pragma unroll
    for (int u = 0; u < 3; ++u)
        A[(size_t)i * T + tid + u * 256] = accj[u] * invB;
}

// degree[i] = sum_j A[i,j];  d_hat[i] = 1/(sqrt(degree)+1e-7)
__global__ __launch_bounds__(256) void degree_kernel(
    const float* __restrict__ A, float* __restrict__ degree,
    float* __restrict__ dhat, int T)
{
    const int i    = blockIdx.x * 8 + (threadIdx.x >> 5);
    const int lane = threadIdx.x & 31;
    if (i >= T) return;
    float s = 0.f;
    for (int j = lane; j < T; j += 32) s += A[(size_t)i * T + j];
    for (int off = 16; off > 0; off >>= 1) s += __shfl_xor(s, off, 32);
    if (lane == 0) {
        degree[i] = s;
        dhat[i]   = 1.f / (sqrtf(s) + 1e-7f);
    }
}

// A_sym (to d_out) and M = I - dhat_i*dhat_j*(degree_i*delta_ij - A_sym)
__global__ __launch_bounds__(256) void asym_M_kernel(
    const float* __restrict__ A, const float* __restrict__ degree,
    const float* __restrict__ dhat, float* __restrict__ Asym,
    float* __restrict__ Mmat, int T)
{
    const size_t idx = (size_t)blockIdx.x * 256 + threadIdx.x;
    if (idx >= (size_t)T * T) return;
    const int i = (int)(idx / T), j = (int)(idx % T);
    const float as = 0.5f * (A[(size_t)i * T + j] + A[(size_t)j * T + i]);
    Asym[idx] = as;
    const float diag = (i == j) ? degree[i] : 0.f;
    Mmat[idx] = ((i == j) ? 1.f : 0.f) - dhat[i] * dhat[j] * (diag - as);
}

__global__ __launch_bounds__(256) void cvt_bf16_kernel(
    const float* __restrict__ in, unsigned short* __restrict__ out, long long n)
{
    const long long i = (long long)blockIdx.x * 256 + threadIdx.x;
    if (i < n) out[i] = f32_to_bf16_rne(in[i]);
}

// out[c*R + r] = bf16(in[r*C + c])  (transpose-convert, R x C input)
__global__ __launch_bounds__(256) void cvt_bf16_T_kernel(
    const float* __restrict__ in, unsigned short* __restrict__ out, int R, int C)
{
    const long long idx = (long long)blockIdx.x * 256 + threadIdx.x;
    if (idx >= (long long)R * C) return;
    const int r = (int)(idx / C), c = (int)(idx % C);
    out[(size_t)c * R + r] = f32_to_bf16_rne(in[idx]);
}

// Mbar = 2*M3 + M2 - 0.5*M
__global__ __launch_bounds__(256) void mbar_kernel(
    const float* __restrict__ M, const float* __restrict__ M2,
    const float* __restrict__ M3, float* __restrict__ Mbar, long long n)
{
    const long long i = (long long)blockIdx.x * 256 + threadIdx.x;
    if (i < n) Mbar[i] = 2.f * M3[i] + M2[i] - 0.5f * M[i];
}

// CW[f,j] = sum_{j'} exp(-2*pi*i*j'*f/T) * w_freq[j',j]
__global__ __launch_bounds__(256) void cw_kernel(
    const float* __restrict__ wf, float* __restrict__ cwre,
    float* __restrict__ cwim, int T, int F)
{
    const int idx = blockIdx.x * 256 + threadIdx.x;
    if (idx >= T * F) return;
    const int f = idx / F, j = idx % F;
    float re = 0.f, im = 0.f;
    for (int jp = 0; jp < F; ++jp) {
        const float th = -2.f * PI_F * (float)jp * (float)f / (float)T;
        float s, c;
        __sincosf(th, &s, &c);
        re += c * wf[jp * F + j];
        im += s * wf[jp * F + j];
    }
    cwre[idx] = re;
    cwim[idx] = im;
}

// G[f,t] = (1/T) * sum_j Re(CW[f,j] * exp(+2*pi*i*j*t/T))
__global__ __launch_bounds__(256) void g_kernel(
    const float* __restrict__ cwre, const float* __restrict__ cwim,
    float* __restrict__ G, int T, int F)
{
    const size_t idx = (size_t)blockIdx.x * 256 + threadIdx.x;
    if (idx >= (size_t)T * T) return;
    const int f = (int)(idx / T), t = (int)(idx % T);
    float acc = 0.f;
    for (int j = 0; j < F; ++j) {
        const float th = 2.f * PI_F * (float)j * (float)t / (float)T;
        float s, c;
        __sincosf(th, &s, &c);
        acc += cwre[f * F + j] * c - cwim[f * F + j] * s;
    }
    G[idx] = acc / (float)T;
}

// out[row] = fc2_b + sum_p part[row*P + p]   (deterministic split-N reduce)
__global__ __launch_bounds__(256) void fc2_reduce_kernel(
    const float* __restrict__ part, const float* __restrict__ b2,
    float* __restrict__ out, int rows, int P)
{
    const int row = blockIdx.x * 256 + threadIdx.x;
    if (row >= rows) return;
    float s = b2[0];
    const float* pr = part + (size_t)row * P;
    for (int p = 0; p < P; ++p) s += pr[p];
    out[row] = s;
}

// ---------------------------------------------------------------------------

extern "C" void kernel_launch(void* const* d_in, const int* in_sizes, int n_in,
                              void* d_out, int out_size, void* d_ws, size_t ws_size,
                              hipStream_t stream)
{
    (void)in_sizes; (void)n_in; (void)out_size; (void)ws_size;
    const int B = 16, T = 768, N = 768, F = 6;
    const long long TT = (long long)T * T;
    const long long BTN = (long long)B * T * N;
    const int P = (T / 64) * 2;                      // 24 partials per row

    const float* x     = (const float*)d_in[0];
    const float* wk    = (const float*)d_in[1];
    const float* wq    = (const float*)d_in[2];
    const float* fc1w  = (const float*)d_in[3];
    const float* fc1b  = (const float*)d_in[4];
    const float* fc2w  = (const float*)d_in[5];
    const float* fc2b  = (const float*)d_in[6];
    const float* wfreq = (const float*)d_in[7];

    float* out_fore = (float*)d_out;                 // (B,1,1,T) = 12288 f32
    float* out_asym = out_fore + (size_t)B * T;      // (T,T)     = 589824 f32

    // workspace carve-out (256B aligned slabs)
    char* wsp = (char*)d_ws;
    auto alloc = [&](size_t bytes) -> void* {
        void* p = (void*)wsp;
        wsp += (bytes + 255) & ~(size_t)255;
        return p;
    };
    float* kbuf   = (float*)alloc((size_t)B * T * 4);
    float* qbuf   = (float*)alloc((size_t)B * T * 4);
    float* Amat   = (float*)alloc(TT * 4);
    float* degv   = (float*)alloc(T * 4);
    float* dhat   = (float*)alloc(T * 4);
    float* Mmat   = (float*)alloc(TT * 4);
    unsigned short* Mbf    = (unsigned short*)alloc(TT * 2);
    unsigned short* MbfT   = (unsigned short*)alloc(TT * 2);
    float* M2mat  = (float*)alloc(TT * 4);
    unsigned short* M2bfT  = (unsigned short*)alloc(TT * 2);
    float* M3mat  = (float*)alloc(TT * 4);
    float* Mbar   = (float*)alloc(TT * 4);
    unsigned short* Mbarbf = (unsigned short*)alloc(TT * 2);
    float* cwre   = (float*)alloc((size_t)T * F * 4);
    float* cwim   = (float*)alloc((size_t)T * F * 4);
    float* Gmat   = (float*)alloc(TT * 4);
    unsigned short* Gbf    = (unsigned short*)alloc(TT * 2);
    unsigned short* fc1T   = (unsigned short*)alloc(TT * 2);
    float* GF     = (float*)alloc(TT * 4);
    unsigned short* GFT    = (unsigned short*)alloc(TT * 2);
    unsigned short* xbf    = (unsigned short*)alloc(BTN * 2);
    unsigned short* Zbf    = (unsigned short*)alloc((size_t)B * TT * 2);
    float* parts  = (float*)alloc((size_t)B * T * P * 4);

    const dim3 blk(256);
    const dim3 gemm_grid(T / 64, T / 128, 1);            // 12 x 6
    const dim3 gemm_gridB(T / 64, T / 128, B);           // 12 x 6 x 16
    const int  rowsBT = B * T;                           // 12288

    // 1. k, q
    kq_kernel<<<rowsBT / 8, blk, 0, stream>>>(x, wk, wq, kbuf, qbuf, rowsBT, N);
    // 2. A = mean_b softmax(scores)
    attn_meanA_kernel<<<T, blk, 0, stream>>>(kbuf, qbuf, Amat, B, T);
    // 3. degree, d_hat
    degree_kernel<<<T / 8, blk, 0, stream>>>(Amat, degv, dhat, T);
    // 4. A_sym (output #2) and M
    asym_M_kernel<<<(unsigned)((TT + 255) / 256), blk, 0, stream>>>(
        Amat, degv, dhat, out_asym, Mmat, T);
    // 5. M -> bf16 (plain + transposed)
    cvt_bf16_kernel<<<(unsigned)((TT + 255) / 256), blk, 0, stream>>>(Mmat, Mbf, TT);
    cvt_bf16_T_kernel<<<(unsigned)((TT + 255) / 256), blk, 0, stream>>>(Mmat, MbfT, T, T);
    // 6. M2 = M @ M  (WMMA)
    gemm_bf16_wmma<<<gemm_grid, blk, 0, stream>>>(Mbf, MbfT, M2mat, nullptr, nullptr,
                                                  T, T, T, 0, 0, 0, nullptr, nullptr, 0);
    // 7. M2 -> bf16 transposed; M3 = M @ M2 (WMMA)
    cvt_bf16_T_kernel<<<(unsigned)((TT + 255) / 256), blk, 0, stream>>>(M2mat, M2bfT, T, T);
    gemm_bf16_wmma<<<gemm_grid, blk, 0, stream>>>(Mbf, M2bfT, M3mat, nullptr, nullptr,
                                                  T, T, T, 0, 0, 0, nullptr, nullptr, 0);
    // 8. Mbar = 2*M3 + M2 - 0.5*M -> bf16
    mbar_kernel<<<(unsigned)((TT + 255) / 256), blk, 0, stream>>>(Mmat, M2mat, M3mat, Mbar, TT);
    cvt_bf16_kernel<<<(unsigned)((TT + 255) / 256), blk, 0, stream>>>(Mbar, Mbarbf, TT);
    // 9. Spectral operator G = Re(DFT6 * w_freq * IDFT6)
    cw_kernel<<<(T * F + 255) / 256, blk, 0, stream>>>(wfreq, cwre, cwim, T, F);
    g_kernel<<<(unsigned)((TT + 255) / 256), blk, 0, stream>>>(cwre, cwim, Gmat, T, F);
    // 10. GF = G @ fc1_w (WMMA)
    cvt_bf16_kernel<<<(unsigned)((TT + 255) / 256), blk, 0, stream>>>(Gmat, Gbf, TT);
    cvt_bf16_T_kernel<<<(unsigned)((TT + 255) / 256), blk, 0, stream>>>(fc1w, fc1T, T, T);
    gemm_bf16_wmma<<<gemm_grid, blk, 0, stream>>>(Gbf, fc1T, GF, nullptr, nullptr,
                                                  T, T, T, 0, 0, 0, nullptr, nullptr, 0);
    cvt_bf16_T_kernel<<<(unsigned)((TT + 255) / 256), blk, 0, stream>>>(GF, GFT, T, T);
    // 11. x -> bf16;  Z[b] = Mbar @ x[b]^T  (Bt == x[b] row-major),
    //     epilogue stores bf16 directly (Z is only ever the next GEMM's A)
    cvt_bf16_kernel<<<(unsigned)((BTN + 255) / 256), blk, 0, stream>>>(x, xbf, BTN);
    gemm_bf16_wmma<<<gemm_gridB, blk, 0, stream>>>(Mbarbf, xbf, nullptr, Zbf, nullptr,
                                                   T, T, T, 0, (long long)T * N, TT,
                                                   nullptr, nullptr, 2);
    // 12. H = leaky(Z @ GF + fc1_b) fused with fc2 dot -> deterministic partials
    gemm_bf16_wmma<<<gemm_gridB, blk, 0, stream>>>(Zbf, GFT, nullptr, nullptr, parts,
                                                   T, T, T, TT, 0, 0,
                                                   fc1b, fc2w, 3);
    // 13. out = fc2_b + sum of 24 per-blockcolumn partials
    fc2_reduce_kernel<<<(rowsBT + 255) / 256, blk, 0, stream>>>(parts, fc2b, out_fore,
                                                                rowsBT, P);
}
